// fullAttention_83708912599314
// MI455X (gfx1250) — compile-verified, matrix operand order checked by reading, not run on hardware
//
#include <hip/hip_runtime.h>
#include <hip/hip_bf16.h>

// Problem constants (match reference)
#define B_   8
#define L1_  2048
#define L2_  2048
#define F_   1024
#define H_   1024
#define NEG_INF_ -1e30f

// Flash-attention tiling
#define QB    32     // queries per workgroup
#define TK    128    // keys per tile
#define LDP   136    // f16 LDS row pitch for P (136 halves = 272B = 17*16B)
#define LDSS  129    // f32 LDS row pitch for S (odd dword pitch -> bank rotation)
#define LDKP  1040   // f16 LDS row pitch for staged K1 (1024 + 2x8-half TDM pads)

typedef __attribute__((ext_vector_type(16))) _Float16 v16h;
typedef __attribute__((ext_vector_type(8)))  _Float16 v8h;
typedef __attribute__((ext_vector_type(8)))  float    v8f;
typedef __attribute__((ext_vector_type(4)))  unsigned int u32x4;
typedef __attribute__((ext_vector_type(8)))  int          i32x8;
typedef __attribute__((ext_vector_type(4)))  int          i32x4;

// ---------------------------------------------------------------------------
// Fragment load: 16x32 f16 tile, "A" layout per CDNA5 ISA 7.12.2.
// Lane L (L<16): row M=L, K = k0+{0..7} -> elems 0-7 and k0+16+{0..7} -> 8-15.
// Lanes 16-31: same rows, K base +8. B operand (B[k,n]=Mat[n,k], Mat row-major
// K-contiguous) uses the lane-symmetric pattern -> same loader for both.
// Works for global or LDS (generic) pointers; needs 16B alignment.
// ---------------------------------------------------------------------------
__device__ __forceinline__ v16h load_frag16(const _Float16* __restrict__ base,
                                            int ld, int row0, int k0) {
  const int lane = threadIdx.x & 31;
  const int r    = row0 + (lane & 15);
  const int ks   = k0 + ((lane >> 4) << 3);
  const _Float16* p = base + (size_t)r * ld + ks;
  v8h lo = *reinterpret_cast<const v8h*>(p);
  v8h hi = *reinterpret_cast<const v8h*>(p + 16);
  v16h out;
#pragma unroll
  for (int i = 0; i < 8; ++i) { out[i] = lo[i]; out[i + 8] = hi[i]; }
  return out;
}

// 64x64 output tile per wave: 4x4 WMMA fragments, f32 accumulation.
template <int KTOT>
__device__ __forceinline__ void gemm_64x64(const _Float16* __restrict__ A, int lda, int row0,
                                           const _Float16* __restrict__ Bm, int ldb, int col0,
                                           v8f acc[4][4]) {
#pragma unroll 1
  for (int k0 = 0; k0 < KTOT; k0 += 32) {
    v16h af[4], bf[4];
#pragma unroll
    for (int i = 0; i < 4; ++i) af[i] = load_frag16(A, lda, row0 + 16 * i, k0);
#pragma unroll
    for (int j = 0; j < 4; ++j) bf[j] = load_frag16(Bm, ldb, col0 + 16 * j, k0);
#pragma unroll
    for (int i = 0; i < 4; ++i)
#pragma unroll
      for (int j = 0; j < 4; ++j)
        acc[i][j] = __builtin_amdgcn_wmma_f32_16x16x32_f16(
            false, af[i], false, bf[j], (short)0, acc[i][j], false, false);
  }
}

// ---------------------------------------------------------------------------
// Elementwise f32 -> f16 convert
// ---------------------------------------------------------------------------
__global__ void k_cvt_f16(const float* __restrict__ x, _Float16* __restrict__ y, int n) {
  int i = blockIdx.x * blockDim.x + threadIdx.x;
  if (i < n) y[i] = (_Float16)x[i];
}

// ---------------------------------------------------------------------------
// x2 [B][L2][H] f32 -> Vt [B][H][L2] f16.  f32 source tile staged into LDS via
// the async global->LDS DMA path (ASYNCcnt-tracked), then convert+transpose.
// ---------------------------------------------------------------------------
__global__ void k_transpose_f16(const float* __restrict__ x, _Float16* __restrict__ y) {
  __shared__ alignas(16) float    ftile[32][32];
  __shared__ alignas(16) _Float16 tile[32][33];
  const int b  = blockIdx.z;
  const int t0 = blockIdx.x * 32;
  const int h0 = blockIdx.y * 32;
  const float* xb = x + (size_t)b * L2_ * H_;
  _Float16* yb    = y + (size_t)b * H_ * L2_;
  const int tid = threadIdx.y * 32 + threadIdx.x;   // 0..255

  {
    const int r  = tid >> 3;
    const int c4 = (tid & 7) << 2;
    const float* gp = xb + (size_t)(t0 + r) * H_ + (h0 + c4);
    unsigned lds_off = (unsigned)(size_t)&ftile[r][c4];
    asm volatile("global_load_async_to_lds_b128 %0, %1, off"
                 :: "v"(lds_off), "v"(gp) : "memory");
  }
  asm volatile("s_wait_asynccnt 0x0" ::: "memory");
  __syncthreads();

#pragma unroll
  for (int i = threadIdx.y; i < 32; i += 8)
    tile[i][threadIdx.x] = (_Float16)ftile[i][threadIdx.x];
  __syncthreads();
#pragma unroll
  for (int i = threadIdx.y; i < 32; i += 8)
    yb[(size_t)(h0 + i) * L2_ + (t0 + threadIdx.x)] = tile[threadIdx.x][i];
}

// ---------------------------------------------------------------------------
// Projection: Y = relu(X @ W^T + b) [* D], f16 out.  X:[M,F], W:[F,F] row-major.
// grid = (M/64, F/256), block = 128 (4 waves, each a 64-wide N slice)
// ---------------------------------------------------------------------------
__global__ void __launch_bounds__(128, 1)
k_proj(const _Float16* __restrict__ X, const _Float16* __restrict__ Wh,
       const float* __restrict__ bias, const float* __restrict__ Dscale,
       _Float16* __restrict__ Y) {
  const int wave = threadIdx.x >> 5;
  const int row0 = blockIdx.x * 64;
  const int col0 = blockIdx.y * 256 + wave * 64;
  v8f acc[4][4] = {};
  gemm_64x64<F_>(X, F_, row0, Wh, F_, col0, acc);

  const int lane = threadIdx.x & 31;
  const int n  = lane & 15;
  const int mb = (lane >> 4) << 3;
#pragma unroll
  for (int j = 0; j < 4; ++j) {
    const int c = col0 + 16 * j + n;
    const float bv = bias[c];
    const float dv = Dscale ? Dscale[c] : 1.0f;
#pragma unroll
    for (int i = 0; i < 4; ++i)
#pragma unroll
      for (int r = 0; r < 8; ++r) {
        float v = acc[i][j][r] + bv;
        v = v > 0.0f ? v : 0.0f;
        Y[(size_t)(row0 + 16 * i + mb + r) * F_ + c] = (_Float16)(v * dv);
      }
  }
}

// ---------------------------------------------------------------------------
// Fused flash attention. Per 32-query block:
//   * K1 query block (32 x F) staged once into LDS via the Tensor Data Mover
//     (TDM, TENSORcnt-tracked) with LDS padding for bank rotation; reused by
//     all 16 key tiles.
//   * Per 128-key tile: S = K1 @ K2^T (8 waves x 2 fragments, A reused),
//     online softmax (running m,l in LDS), P = exp(S-m) as f16 in LDS,
//     acc = acc*corr + P @ V (each wave owns a 128-wide H slice).
// grid = (L1/32, B), block = 256 (8 waves)
// ---------------------------------------------------------------------------
__global__ void __launch_bounds__(256, 1)
k_flash(const _Float16* __restrict__ K1, const _Float16* __restrict__ K2,
        const _Float16* __restrict__ Vt, const unsigned char* __restrict__ mask,
        float* __restrict__ out) {
  __shared__ alignas(16) _Float16 K1ld[QB * LDKP];   // 66.5 KB staged K1 block
  __shared__ alignas(16) float    Sld[QB][LDSS];     // 16.5 KB
  __shared__ alignas(16) _Float16 Pld[QB][LDP];      //  8.7 KB
  __shared__ float m_row[QB], l_row[QB], corr_row[QB];
  __shared__ float pmax[QB][8], psum[QB][8];

  const int b  = blockIdx.y;
  const int q0 = blockIdx.x * QB;
  const _Float16* K1b = K1 + (size_t)b * L1_ * F_;
  const _Float16* K2b = K2 + (size_t)b * L2_ * F_;
  const _Float16* Vtb = Vt + (size_t)b * H_ * L2_;
  const unsigned char* mkb = mask + (size_t)b * L2_;

  const int tid  = threadIdx.x;
  const int wave = tid >> 5;
  const int lane = tid & 31;
  const int mi   = wave & 1;          // S fragment row group
  const int tj0  = (wave >> 1) << 1;  // S fragment col groups tj0, tj0+1
  const int h0   = wave * 128;        // PV H slice owned by this wave
  const int n    = lane & 15;
  const int mb   = (lane >> 4) << 3;
  const int srow = tid & 31;          // stats mapping: row
  const int schk = tid >> 5;          // stats mapping: 16-col chunk

  if (tid < QB) { m_row[tid] = -3.0e38f; l_row[tid] = 0.0f; }

  // ---- Stage K1 query block into LDS (padded pitch LDKP = 1024 + 2x8 halves)
#if __has_builtin(__builtin_amdgcn_tensor_load_to_lds) && \
    __has_builtin(__builtin_amdgcn_s_wait_tensorcnt)
  if (wave == 0) {
    const unsigned long long ga =
        (unsigned long long)(size_t)(K1b + (size_t)q0 * F_);
    u32x4 g0;
    g0[0] = 1u;                                   // count=1, user descriptor
    g0[1] = (unsigned)(size_t)&K1ld[0];           // lds_addr
    g0[2] = (unsigned)(ga & 0xFFFFFFFFu);         // global_addr[31:0]
    g0[3] = (unsigned)((ga >> 32) & 0x01FFFFFFu)  // global_addr[56:32]
          | (2u << 30);                           // type = 2 ("image")
    i32x8 g1;
    g1[0] = (1 << 16)      // data_size = 2 bytes
          | (1 << 20)      // pad_enable
          | (7 << 22)      // pad_interval: every 256 dwords (512 halves)
          | (3 << 25);     // pad_amount: 4 dwords (8 halves)
    g1[1] = (int)(1024u << 16);   // tensor_dim0[15:0] = 1024
    g1[2] = (int)(32u << 16);     // tensor_dim0 hi | tensor_dim1[15:0] = 32
    g1[3] = (int)(1024u << 16);   // tensor_dim1 hi | tile_dim0 = 1024
    g1[4] = 32;                   // tile_dim1 = 32, tile_dim2 = 0
    g1[5] = 1024;                 // tensor_dim0_stride[31:0] = F
    g1[6] = 0;
    g1[7] = 0;
    i32x4 g2 = {0, 0, 0, 0};      // dims 3/4 unused
    i32x4 g3 = {0, 0, 0, 0};
    i32x8 g4 = {0, 0, 0, 0, 0, 0, 0, 0};  // trailing group (6-arg toolchain form)
    __builtin_amdgcn_tensor_load_to_lds(g0, g1, g2, g3, g4, 0);
    __builtin_amdgcn_s_wait_tensorcnt((short)0);
  }
#else
  // Fallback: cooperative b128 copies producing the identical padded layout.
  for (int idx = tid; idx < QB * 128; idx += 256) {
    const int row = idx >> 7;
    const int c   = idx & 127;                    // 8-half chunk within row
    const _Float16* sp = K1b + (size_t)(q0 + row) * F_ + c * 8;
    _Float16* dp = &K1ld[row * LDKP + c * 8 + (c >= 64 ? 8 : 0)];
    *reinterpret_cast<v8h*>(dp) = *reinterpret_cast<const v8h*>(sp);
  }
#endif
  v8f acc[2][8] = {};               // 32 x 128 f32 accumulators
  __syncthreads();

#pragma unroll 1
  for (int t0 = 0; t0 < L2_; t0 += TK) {
    // ---- Phase A: two 16x16 S fragments per wave, A fragment from LDS
    v8f sacc0 = {}, sacc1 = {};
#pragma unroll 1
    for (int k0 = 0; k0 < F_; k0 += 32) {
      const int kk = k0 + (k0 >= 512 ? 8 : 0);    // skip TDM pad halves
      v16h a  = load_frag16(K1ld, LDKP, 16 * mi, kk);
      v16h b0 = load_frag16(K2b, F_, t0 + 16 * tj0, k0);
      v16h b1 = load_frag16(K2b, F_, t0 + 16 * (tj0 + 1), k0);
      if (k0 + 128 < F_) {   // speculative L2 prefetch of upcoming K step
        __builtin_prefetch(K2b + (size_t)(t0 + 16 * tj0 + n) * F_ + k0 + 128, 0, 1);
        __builtin_prefetch(K2b + (size_t)(t0 + 16 * (tj0 + 1) + n) * F_ + k0 + 128, 0, 1);
      }
      sacc0 = __builtin_amdgcn_wmma_f32_16x16x32_f16(false, a, false, b0,
                                                     (short)0, sacc0, false, false);
      sacc1 = __builtin_amdgcn_wmma_f32_16x16x32_f16(false, a, false, b1,
                                                     (short)0, sacc1, false, false);
    }
#pragma unroll
    for (int r = 0; r < 8; ++r) {
      Sld[16 * mi + mb + r][16 * tj0 + n]        = sacc0[r];
      Sld[16 * mi + mb + r][16 * (tj0 + 1) + n]  = sacc1[r];
    }
    __syncthreads();

    // ---- Phase B1: per-row tile max (masked)
    {
      float mx = -3.0e38f;
#pragma unroll
      for (int c = 0; c < 16; ++c) {
        const int col = schk * 16 + c;
        float v = Sld[srow][col];
        if (mkb[t0 + col]) v = NEG_INF_;
        mx = fmaxf(mx, v);
      }
      pmax[srow][schk] = mx;
    }
    __syncthreads();
    if (tid < QB) {
      float mx = pmax[tid][0];
#pragma unroll
      for (int c = 1; c < 8; ++c) mx = fmaxf(mx, pmax[tid][c]);
      const float mo = m_row[tid];
      const float mn = fmaxf(mo, mx);
      corr_row[tid] = __expf(mo - mn);
      m_row[tid] = mn;
    }
    __syncthreads();

    // ---- Phase B2: P = exp(S - m) as f16, per-row partial sums
    {
      const float mn = m_row[srow];
      float s = 0.0f;
#pragma unroll
      for (int c = 0; c < 16; ++c) {
        const int col = schk * 16 + c;
        float v = Sld[srow][col];
        if (mkb[t0 + col]) v = NEG_INF_;
        const float e = __expf(v - mn);
        Pld[srow][col] = (_Float16)e;
        s += e;
      }
      psum[srow][schk] = s;
    }
    __syncthreads();
    if (tid < QB) {
      float s = psum[tid][0];
#pragma unroll
      for (int c = 1; c < 8; ++c) s += psum[tid][c];
      l_row[tid] = l_row[tid] * corr_row[tid] + s;
    }

    // ---- Phase C: rescale accumulators, then acc += P @ V(tile, H-slice)
    {
      float cr[2][8];
#pragma unroll
      for (int i = 0; i < 2; ++i)
#pragma unroll
        for (int r = 0; r < 8; ++r)
          cr[i][r] = corr_row[16 * i + mb + r];
#pragma unroll
      for (int i = 0; i < 2; ++i)
#pragma unroll
        for (int j = 0; j < 8; ++j)
#pragma unroll
          for (int r = 0; r < 8; ++r)
            acc[i][j][r] *= cr[i][r];
    }
#pragma unroll 1
    for (int ks = 0; ks < TK; ks += 32) {
      v16h aP[2];
#pragma unroll
      for (int i = 0; i < 2; ++i)
        aP[i] = load_frag16(&Pld[0][0], LDP, 16 * i, ks);   // ds_load_b128 x2
#pragma unroll
      for (int j = 0; j < 8; ++j) {
        v16h bV = load_frag16(Vtb, L2_, h0 + 16 * j, t0 + ks);
#pragma unroll
        for (int i = 0; i < 2; ++i)
          acc[i][j] = __builtin_amdgcn_wmma_f32_16x16x32_f16(
              false, aP[i], false, bV, (short)0, acc[i][j], false, false);
      }
    }
    __syncthreads();   // Sld/Pld reused next tile; also covers l_row update
  }

  // ---- Epilogue: out = acc / l
  float* ob = out + (size_t)b * L1_ * H_;
#pragma unroll
  for (int i = 0; i < 2; ++i)
#pragma unroll
    for (int r = 0; r < 8; ++r) {
      const int row = 16 * i + mb + r;
      const float inv = 1.0f / l_row[row];
#pragma unroll
      for (int j = 0; j < 8; ++j)
        ob[(size_t)(q0 + row) * H_ + (h0 + 16 * j + n)] = acc[i][j][r] * inv;
    }
}

// ---------------------------------------------------------------------------
// Host-side orchestration
// ---------------------------------------------------------------------------
extern "C" void kernel_launch(void* const* d_in, const int* in_sizes, int n_in,
                              void* d_out, int out_size, void* d_ws, size_t ws_size,
                              hipStream_t stream) {
  const float* xh1 = (const float*)d_in[0];                     // [B,L1,F]
  const float* xh2 = (const float*)d_in[1];                     // [B,L2,F]
  const float* x2  = (const float*)d_in[2];                     // [B,L2,H]
  const unsigned char* x2_mask = (const unsigned char*)d_in[3]; // [B,L2] bool
  const float* W   = (const float*)d_in[4];                     // [F,F]
  const float* bia = (const float*)d_in[5];                     // [F]
  const float* D   = (const float*)d_in[6];                     // [1,F]
  float* out = (float*)d_out;                                   // [B,L1,H]

  // Workspace layout (162 MiB total)
  char* base = (char*)d_ws;
  const size_t SZ_Wh = (size_t)F_ * F_ * sizeof(_Float16);        //   2 MiB
  const size_t SZ_Xh = (size_t)B_ * L1_ * F_ * sizeof(_Float16);  //  32 MiB
  const size_t SZ_Vt = (size_t)B_ * H_ * L2_ * sizeof(_Float16);  //  32 MiB
  _Float16* Wh  = (_Float16*)(base);
  _Float16* X1h = (_Float16*)(base + SZ_Wh);
  _Float16* X2h = (_Float16*)(base + SZ_Wh + SZ_Xh);
  _Float16* VtB = (_Float16*)(base + SZ_Wh + 2 * SZ_Xh);
  _Float16* K1  = (_Float16*)(base + SZ_Wh + 2 * SZ_Xh + SZ_Vt);
  _Float16* K2  = (_Float16*)(base + SZ_Wh + 2 * SZ_Xh + SZ_Vt + SZ_Xh);

  // 1) down-convert inputs to f16
  {
    int n = F_ * F_;
    k_cvt_f16<<<(n + 255) / 256, 256, 0, stream>>>(W, Wh, n);
    n = B_ * L1_ * F_;
    k_cvt_f16<<<(n + 255) / 256, 256, 0, stream>>>(xh1, X1h, n);
    n = B_ * L2_ * F_;
    k_cvt_f16<<<(n + 255) / 256, 256, 0, stream>>>(xh2, X2h, n);
  }
  // x2 -> Vt (transposed f16, staged through LDS via async DMA)
  {
    dim3 g(L2_ / 32, H_ / 32, B_), blk(32, 8);
    k_transpose_f16<<<g, blk, 0, stream>>>(x2, VtB);
  }
  // 2) shared projection GEMMs
  {
    dim3 g(B_ * L1_ / 64, F_ / 256), blk(128);
    k_proj<<<g, blk, 0, stream>>>(X1h, Wh, bia, nullptr, K1);
    k_proj<<<g, blk, 0, stream>>>(X2h, Wh, bia, D, K2);
  }
  // 3) fused scores + softmax + PV (TDM-staged K1 block)
  {
    dim3 g(L1_ / QB, B_), blk(256);
    k_flash<<<g, blk, 0, stream>>>(K1, K2, VtB, x2_mask, out);
  }
}
